// MSGATLayer_3307124818060
// MI455X (gfx1250) — matrix operand order repaired and checked by reading, not compiled, not run on hardware
//
#include <hip/hip_runtime.h>
#include <math.h>

#define NN   50000
#define INF_ 128
#define MM   3
#define EE   800000
#define HIDN 128
#define SLOPE 0.2f
#define EPSLN 1e-5f

typedef __attribute__((ext_vector_type(16))) _Float16 v16h;
typedef __attribute__((ext_vector_type(8)))  _Float16 v8h;
typedef __attribute__((ext_vector_type(8)))  float    v8f;

__device__ inline float waveSum(float v) {
#pragma unroll
  for (int off = 16; off > 0; off >>= 1) v += __shfl_xor(v, off, 32);
  return v;
}

__device__ inline void atomicMaxF(float* addr, float v) {
  if (v >= 0.0f) atomicMax((int*)addr, __float_as_int(v));
  else           atomicMin((unsigned int*)addr, __float_as_uint(v));
}

// ---------------- conversion / small precompute ----------------
__global__ void k_f32_to_f16(_Float16* dst, const float* src, long n) {
  long i = (long)blockIdx.x * blockDim.x + threadIdx.x;
  if (i < n) dst[i] = (_Float16)src[i];
}

// aw[m][t][k], t=0..3 -> el head t (a_l,W_src); t=4..7 -> er head t-4 (a_r,W_dst)
__global__ void k_combine_aw(float* aw, const float* a_l, const float* a_r,
                             const float* W_src, const float* W_dst) {
  int tid = blockIdx.x * blockDim.x + threadIdx.x;
  if (tid >= MM * 8 * 128) return;
  int k = tid & 127, t = (tid >> 7) & 7, m = tid >> 10;
  int h = t & 3;
  const float* av = (t < 4 ? a_l : a_r) + ((size_t)m * 4 + h) * 32;
  const float* Wv = (t < 4 ? W_src : W_dst) + (size_t)m * 128 * 128;
  float s = 0.0f;
#pragma unroll 4
  for (int o = 0; o < 32; ++o) s += av[o] * Wv[(size_t)(h * 32 + o) * 128 + k];
  aw[((size_t)m * 8 + t) * 128 + k] = s;
}

// el[m][n][h], er[m][n][h] = h_row . aw_vec   (pure f32 path)
__global__ void k_el_er(float* el, float* er, const float* h, const float* aw) {
  long tid = (long)blockIdx.x * blockDim.x + threadIdx.x;
  if (tid >= (long)MM * NN * 8) return;
  int t = (int)(tid & 7);
  long nm = tid >> 3;
  int n = (int)(nm % NN), m = (int)(nm / NN);
  const float* hr = h + (size_t)n * INF_;
  const float* ar = aw + ((size_t)m * 8 + t) * 128;
  float s = 0.0f;
#pragma unroll 8
  for (int k = 0; k < 128; ++k) s += hr[k] * ar[k];
  if (t < 4) el[((size_t)m * NN + n) * 4 + t]       = s;
  else       er[((size_t)m * NN + n) * 4 + (t - 4)] = s;
}

// ---------------- per-metapath init ----------------
__global__ void k_init_m(float* zm, float* emax, float* den) {
  long i = (long)blockIdx.x * blockDim.x + threadIdx.x;
  if (i >= (long)NN * 128) return;
  zm[i] = 0.0f;
  if (i < (long)NN * 4) { emax[i] = -INFINITY; den[i] = 0.0f; }
}

// ---------------- WMMA GEMM: fs = h16 @ W16[m]^T  (N x 128) ----------------
__global__ void k_gemm_fs(float* fs, const _Float16* h16, const _Float16* w16m) {
  __shared__ __align__(16) _Float16 As[16 * 128];   // 16-row A panel, 4 KB
  int tid  = threadIdx.x;
  int wv   = tid >> 5, lane = tid & 31;
  int rowBase = blockIdx.x * 16;
  // stage A panel (each thread: 8 halves = one float4)
  ((float4*)As)[tid] = ((const float4*)(h16 + (size_t)rowBase * 128))[tid];
  __syncthreads();

  int colBase = wv * 16;
  int r  = lane & 15;
  int hi = lane >> 4;

  v8f c = {};
#pragma unroll
  for (int kk = 0; kk < 4; ++kk) {
    int k0 = kk * 32;
    // A fragment (16x32 f16, contiguous K): lane<16 holds K k0..k0+15
    const _Float16* Ap = As + r * 128 + k0 + (hi ? 16 : 0);
    v8h alo = *(const v8h*)Ap;
    v8h ahi = *(const v8h*)(Ap + 8);
    v16h a;
#pragma unroll
    for (int i = 0; i < 8; ++i) { a[i] = alo[i]; a[8 + i] = ahi[i]; }
    // B fragment (32x16): B[k][n] = W[colBase+n][k]; lane<16 -> K k0..k0+15
    const _Float16* Bp = w16m + (size_t)(colBase + r) * 128 + k0 + (hi ? 16 : 0);
    v16h b = *(const v16h*)Bp;
    c = __builtin_amdgcn_wmma_f32_16x16x32_f16(false, a, false, b,
                                               (short)0, c, false, false);
  }
  // D layout: lane<16 -> M=vgpr, lane>=16 -> M=8+vgpr; N = lane&15
  int col = colBase + r;
  int rowOff = hi ? 8 : 0;
#pragma unroll
  for (int i = 0; i < 8; ++i)
    fs[(size_t)(rowBase + rowOff + i) * 128 + col] = c[i];
}

// ---------------- edge softmax (3 passes) ----------------
__global__ void k_edge_max(const float* elm, const float* erm, float* emax,
                           const int* srcm, const int* dstm) {
  long e = (long)blockIdx.x * blockDim.x + threadIdx.x;
  if (e >= EE) return;
  int s = srcm[e], d = dstm[e];
  float4 a = ((const float4*)elm)[s];
  float4 b = ((const float4*)erm)[d];
  float v0 = a.x + b.x, v1 = a.y + b.y, v2 = a.z + b.z, v3 = a.w + b.w;
  v0 = v0 > 0.f ? v0 : SLOPE * v0;  v1 = v1 > 0.f ? v1 : SLOPE * v1;
  v2 = v2 > 0.f ? v2 : SLOPE * v2;  v3 = v3 > 0.f ? v3 : SLOPE * v3;
  atomicMaxF(&emax[(size_t)d * 4 + 0], v0);
  atomicMaxF(&emax[(size_t)d * 4 + 1], v1);
  atomicMaxF(&emax[(size_t)d * 4 + 2], v2);
  atomicMaxF(&emax[(size_t)d * 4 + 3], v3);
}

__global__ void k_edge_den(const float* elm, const float* erm, const float* emax,
                           float* den, const int* srcm, const int* dstm) {
  long e = (long)blockIdx.x * blockDim.x + threadIdx.x;
  if (e >= EE) return;
  int s = srcm[e], d = dstm[e];
  float4 a = ((const float4*)elm)[s];
  float4 b = ((const float4*)erm)[d];
  float4 mx = ((const float4*)emax)[d];
  float v0 = a.x + b.x, v1 = a.y + b.y, v2 = a.z + b.z, v3 = a.w + b.w;
  v0 = v0 > 0.f ? v0 : SLOPE * v0;  v1 = v1 > 0.f ? v1 : SLOPE * v1;
  v2 = v2 > 0.f ? v2 : SLOPE * v2;  v3 = v3 > 0.f ? v3 : SLOPE * v3;
  atomicAdd(&den[(size_t)d * 4 + 0], expf(v0 - mx.x));
  atomicAdd(&den[(size_t)d * 4 + 1], expf(v1 - mx.y));
  atomicAdd(&den[(size_t)d * 4 + 2], expf(v2 - mx.z));
  atomicAdd(&den[(size_t)d * 4 + 3], expf(v3 - mx.w));
}

// one wave per edge: lane covers 4 features; head = lane>>3
__global__ void k_edge_aggr(float* zm, const float* fs,
                            const float* elm, const float* erm,
                            const float* emax, const float* den,
                            const int* srcm, const int* dstm) {
  int wv = threadIdx.x >> 5, lane = threadIdx.x & 31;
  long e = (long)blockIdx.x * 8 + wv;
  if (e >= EE) return;
  int s = srcm[e], d = dstm[e];
  int h = lane >> 3;
  float ev = elm[(size_t)s * 4 + h] + erm[(size_t)d * 4 + h];
  ev = ev > 0.f ? ev : SLOPE * ev;
  float alpha = expf(ev - emax[(size_t)d * 4 + h]) / den[(size_t)d * 4 + h];
  float4 f = ((const float4*)fs)[(size_t)s * 32 + lane];
  float* zd = zm + (size_t)d * INF_ + lane * 4;
  atomicAdd(zd + 0, f.x * alpha);
  atomicAdd(zd + 1, f.y * alpha);
  atomicAdd(zd + 2, f.z * alpha);
  atomicAdd(zd + 3, f.w * alpha);
}

__global__ void k_bias_elu(float* zm, const float* bias_m) {
  long i = (long)blockIdx.x * blockDim.x + threadIdx.x;
  if (i >= (long)NN * 128) return;
  float v = zm[i] + bias_m[i & 127];
  zm[i] = v > 0.f ? v : expm1f(v);
}

// ---------------- semantic attention ----------------
__global__ void k_zero_wsum(float* wsum) {
  if (threadIdx.x < MM) wsum[threadIdx.x] = 0.0f;
}

// one wave per (m,n): w = tanh(z_row @ p1_w^T + p1_b) . p2_w ; atomic into wsum[m]
__global__ void k_semantic(const float* z, const float* p1w, const float* p1b,
                           const float* p2w, float* wsum) {
  __shared__ __align__(16) float zrow[8][128];
  int wv = threadIdx.x >> 5, lane = threadIdx.x & 31;
  long idx = (long)blockIdx.x * 8 + wv;
  if (idx >= (long)MM * NN) return;
  int m = (int)(idx / NN), n = (int)(idx % NN);
  const float* zr = z + ((size_t)m * NN + n) * INF_;
  ((float4*)&zrow[wv][0])[lane] = ((const float4*)zr)[lane];
  __syncthreads();
  float acc = 0.0f;
#pragma unroll
  for (int jj = 0; jj < 4; ++jj) {
    int j = lane + jj * 32;
    const float* pr = p1w + (size_t)j * 128;
    float t = p1b[j];
#pragma unroll 8
    for (int k = 0; k < 128; ++k) t += zrow[wv][k] * pr[k];
    acc += tanhf(t) * p2w[j];
  }
  acc = waveSum(acc);
  if (lane == 0) atomicAdd(&wsum[m], acc);
}

__global__ void k_beta(const float* wsum, float* beta) {
  if (threadIdx.x != 0 || blockIdx.x != 0) return;
  float w0 = wsum[0] / (float)NN, w1 = wsum[1] / (float)NN, w2 = wsum[2] / (float)NN;
  float mx = fmaxf(w0, fmaxf(w1, w2));
  float e0 = expf(w0 - mx), e1 = expf(w1 - mx), e2 = expf(w2 - mx);
  float s = e0 + e1 + e2;
  beta[0] = e0 / s; beta[1] = e1 / s; beta[2] = e2 / s;
}

// ---------------- output: layernorm(h) + sum_m beta_m * z_m ----------------
__global__ void k_final(float* out, const float* h, const float* z,
                        const float* beta, const float* lnw, const float* lnb) {
  int wv = threadIdx.x >> 5, lane = threadIdx.x & 31;
  int n = blockIdx.x * 8 + wv;
  if (n >= NN) return;
  float4 x = ((const float4*)(h + (size_t)n * INF_))[lane];
  float s  = x.x + x.y + x.z + x.w;
  float sq = x.x * x.x + x.y * x.y + x.z * x.z + x.w * x.w;
  s = waveSum(s); sq = waveSum(sq);
  float mu  = s * (1.0f / 128.0f);
  float var = sq * (1.0f / 128.0f) - mu * mu;
  float inv = rsqrtf(var + EPSLN);
  float b0 = beta[0], b1 = beta[1], b2 = beta[2];
  size_t base = (size_t)n * INF_ + lane * 4;
  float xs[4] = {x.x, x.y, x.z, x.w};
#pragma unroll
  for (int i = 0; i < 4; ++i) {
    int f = lane * 4 + i;
    float hn = (xs[i] - mu) * inv * lnw[f] + lnb[f];
    float sem = b0 * z[base + i]
              + b1 * z[(size_t)NN * 128 + base + i]
              + b2 * z[(size_t)2 * NN * 128 + base + i];
    out[base + i] = hn + sem;
  }
}

// ---------------- host: launch sequence ----------------
extern "C" void kernel_launch(void* const* d_in, const int* in_sizes, int n_in,
                              void* d_out, int out_size, void* d_ws, size_t ws_size,
                              hipStream_t stream) {
  const float* h     = (const float*)d_in[0];
  const float* W_src = (const float*)d_in[1];
  const float* W_dst = (const float*)d_in[2];
  const float* a_l   = (const float*)d_in[3];
  const float* a_r   = (const float*)d_in[4];
  const float* bias  = (const float*)d_in[5];
  const float* p1_w  = (const float*)d_in[6];
  const float* p1_b  = (const float*)d_in[7];
  const float* p2_w  = (const float*)d_in[8];
  const float* ln_w  = (const float*)d_in[9];
  const float* ln_b  = (const float*)d_in[10];
  const int*   src   = (const int*)d_in[11];
  const int*   dst   = (const int*)d_in[12];
  float* out = (float*)d_out;

  char* ws = (char*)d_ws;
  constexpr size_t OFF_H16  = 0;                                   // N*128 f16
  constexpr size_t OFF_W16  = OFF_H16  + (size_t)NN * 128 * 2;     // M*128*128 f16
  constexpr size_t OFF_AW   = OFF_W16  + (size_t)MM * 128 * 128 * 2;
  constexpr size_t OFF_EL   = OFF_AW   + (size_t)MM * 8 * 128 * 4;
  constexpr size_t OFF_ER   = OFF_EL   + (size_t)MM * NN * 4 * 4;
  constexpr size_t OFF_FS   = OFF_ER   + (size_t)MM * NN * 4 * 4;  // N*128 f32 (reused per m)
  constexpr size_t OFF_EMAX = OFF_FS   + (size_t)NN * 128 * 4;
  constexpr size_t OFF_DEN  = OFF_EMAX + (size_t)NN * 4 * 4;
  constexpr size_t OFF_Z    = OFF_DEN  + (size_t)NN * 4 * 4;       // M*N*128 f32
  constexpr size_t OFF_WSUM = OFF_Z    + (size_t)MM * NN * 128 * 4;
  constexpr size_t OFF_BETA = OFF_WSUM + 256;

  _Float16* h16  = (_Float16*)(ws + OFF_H16);
  _Float16* W16  = (_Float16*)(ws + OFF_W16);
  float* aw   = (float*)(ws + OFF_AW);
  float* el   = (float*)(ws + OFF_EL);
  float* er   = (float*)(ws + OFF_ER);
  float* fs   = (float*)(ws + OFF_FS);
  float* emax = (float*)(ws + OFF_EMAX);
  float* den  = (float*)(ws + OFF_DEN);
  float* z    = (float*)(ws + OFF_Z);
  float* wsum = (float*)(ws + OFF_WSUM);
  float* beta = (float*)(ws + OFF_BETA);

  const int T = 256;
  // precompute
  k_f32_to_f16<<<(int)(((long)NN * 128 + T - 1) / T), T, 0, stream>>>(h16, h, (long)NN * 128);
  k_f32_to_f16<<<(int)(((long)MM * 128 * 128 + T - 1) / T), T, 0, stream>>>(W16, W_src, (long)MM * 128 * 128);
  k_combine_aw<<<(MM * 8 * 128 + T - 1) / T, T, 0, stream>>>(aw, a_l, a_r, W_src, W_dst);
  k_el_er<<<(int)(((long)MM * NN * 8 + T - 1) / T), T, 0, stream>>>(el, er, h, aw);
  k_zero_wsum<<<1, 32, 0, stream>>>(wsum);

  for (int m = 0; m < MM; ++m) {
    float* zm = z + (size_t)m * NN * 128;
    const float* elm = el + (size_t)m * NN * 4;
    const float* erm = er + (size_t)m * NN * 4;
    const int* srcm = src + (size_t)m * EE;
    const int* dstm = dst + (size_t)m * EE;

    k_init_m<<<(int)(((long)NN * 128 + T - 1) / T), T, 0, stream>>>(zm, emax, den);
    k_gemm_fs<<<NN / 16, T, 0, stream>>>(fs, h16, W16 + (size_t)m * 128 * 128);
    k_edge_max<<<(EE + T - 1) / T, T, 0, stream>>>(elm, erm, emax, srcm, dstm);
    k_edge_den<<<(EE + T - 1) / T, T, 0, stream>>>(elm, erm, emax, den, srcm, dstm);
    k_edge_aggr<<<EE / 8, T, 0, stream>>>(zm, fs, elm, erm, emax, den, srcm, dstm);
    k_bias_elu<<<(int)(((long)NN * 128 + T - 1) / T), T, 0, stream>>>(zm, bias + (size_t)m * 128);
  }

  k_semantic<<<(int)(((long)MM * NN + 7) / 8), T, 0, stream>>>(z, p1_w, p1_b, p2_w, wsum);
  k_beta<<<1, 1, 0, stream>>>(wsum, beta);
  k_final<<<(NN + 7) / 8, T, 0, stream>>>(out, h, z, beta, ln_w, ln_b);
}
